// Mamba2Model_30064771072276
// MI455X (gfx1250) — compile-verified, matrix operand order checked by reading, not run on hardware
//
#include <hip/hip_runtime.h>
#include <hip/hip_bf16.h>
#include <math.h>

// ---------------------------------------------------------------------------
// Mamba2 classifier head for MI455X (gfx1250, wave32, WMMA).
// Only hidden[:, -1, :] feeds the output, so z / C / out_proj / per-step y
// are pruned to the final timestep. Pipeline:
//   k_inproj_gemm : u[10976x192] @ W[1536:3520]^T -> xBC|dt  (WMMA bf16)
//   k_zlast       : z at final timestep (tiny dot products)
//   k_conv_dt     : causal depthwise conv(k=4)+SiLU; softplus(dt)+dA
//   k_scan        : 768 independent (b,h) chains, 8x128 fp32 state in VGPRs
//   k_norm        : gate with silu(z), RMSNorm -> bf16, pad M to 16
//   k_cls         : y[16x1536] @ cls_w^T (WMMA bf16) + bias, broadcast to T
// ---------------------------------------------------------------------------

typedef __bf16 bf16_t;
typedef bf16_t v16bf __attribute__((ext_vector_type(16)));
typedef bf16_t v8bf  __attribute__((ext_vector_type(8)));
typedef float  v8f   __attribute__((ext_vector_type(8)));

#define NB_     4
#define NT_     14
#define NTOK    197
#define DMODEL  192
#define DINNER  1536
#define DSTATE  128
#define NHEADS  192
#define CONVDIM 1792                 // D_INNER + 2*D_STATE
#define NCOLS   1984                 // CONVDIM + NHEADS (xBC + dt)
#define LSEQ    2744                 // 14 * 196
#define MROWS   (NB_ * LSEQ)         // 10976
#define NUMCLASS 2513
#define NCLS_TILES 158               // ceil(2513/16)

// Build a 16x32 bf16 WMMA fragment half-row from an fp32 row pointer.
// Per ISA layout: half j -> K = kb + (j>=8)*16 + (j&7) + hi*8.
__device__ __forceinline__ v16bf frag_from_f32(const float* __restrict__ row,
                                               int kb, int hi) {
  const float4 q0 = *(const float4*)(row + kb + hi * 8);
  const float4 q1 = *(const float4*)(row + kb + hi * 8 + 4);
  const float4 q2 = *(const float4*)(row + kb + 16 + hi * 8);
  const float4 q3 = *(const float4*)(row + kb + 20 + hi * 8);
  v16bf f;
  f[0] = (bf16_t)q0.x;  f[1] = (bf16_t)q0.y;  f[2] = (bf16_t)q0.z;  f[3] = (bf16_t)q0.w;
  f[4] = (bf16_t)q1.x;  f[5] = (bf16_t)q1.y;  f[6] = (bf16_t)q1.z;  f[7] = (bf16_t)q1.w;
  f[8] = (bf16_t)q2.x;  f[9] = (bf16_t)q2.y;  f[10] = (bf16_t)q2.z; f[11] = (bf16_t)q2.w;
  f[12] = (bf16_t)q3.x; f[13] = (bf16_t)q3.y; f[14] = (bf16_t)q3.z; f[15] = (bf16_t)q3.w;
  return f;
}

// ---------------------------------------------------------------------------
// Kernel 1: in_proj GEMM. One wave -> 16x64 strip. K = 192 (6 x 32 bf16).
// A rows gather u = inputs[:, :, 1:, :] (drop CLS token per frame).
// B rows are W[1536 + n] (xBC and dt columns are contiguous in W).
// ---------------------------------------------------------------------------
__global__ __launch_bounds__(128) void k_inproj_gemm(
    const float* __restrict__ inp, const float* __restrict__ W,
    bf16_t* __restrict__ ws1) {
  const int lane = threadIdx.x & 31;
  const int wid  = blockIdx.x * 4 + (threadIdx.x >> 5);
  const int NT   = NCOLS / 64;                      // 31
  if (wid >= (MROWS / 16) * NT) return;
  const int m0 = (wid / NT) * 16;
  const int n0 = (wid % NT) * 64;
  const int lo = lane & 15, hi = lane >> 4;

  // A-row gather: m -> inputs[b][tt][tok][.] with tok = (m%196)+1
  const int m  = m0 + lo;
  const int b  = m / LSEQ, l = m % LSEQ;
  const int tt = l / 196,  tk = (l % 196) + 1;
  const float* __restrict__ arow =
      inp + ((size_t)(b * NT_ + tt) * NTOK + tk) * DMODEL;

  const float* __restrict__ brow0 = W + (size_t)(1536 + n0 +  0 + lo) * DMODEL;
  const float* __restrict__ brow1 = W + (size_t)(1536 + n0 + 16 + lo) * DMODEL;
  const float* __restrict__ brow2 = W + (size_t)(1536 + n0 + 32 + lo) * DMODEL;
  const float* __restrict__ brow3 = W + (size_t)(1536 + n0 + 48 + lo) * DMODEL;

  v8f acc0 = {}, acc1 = {}, acc2 = {}, acc3 = {};
#pragma unroll
  for (int ks = 0; ks < DMODEL / 32; ++ks) {
    const int kb = ks * 32;
    const v16bf a  = frag_from_f32(arow,  kb, hi);
    const v16bf b0 = frag_from_f32(brow0, kb, hi);
    const v16bf b1 = frag_from_f32(brow1, kb, hi);
    const v16bf b2 = frag_from_f32(brow2, kb, hi);
    const v16bf b3 = frag_from_f32(brow3, kb, hi);
    acc0 = __builtin_amdgcn_wmma_f32_16x16x32_bf16(false, a, false, b0, (short)0, acc0, false, false);
    acc1 = __builtin_amdgcn_wmma_f32_16x16x32_bf16(false, a, false, b1, (short)0, acc1, false, false);
    acc2 = __builtin_amdgcn_wmma_f32_16x16x32_bf16(false, a, false, b2, (short)0, acc2, false, false);
    acc3 = __builtin_amdgcn_wmma_f32_16x16x32_bf16(false, a, false, b3, (short)0, acc3, false, false);
  }
  // C/D layout: VGPR r -> row m0 + r + hi*8, lane lo -> col
#pragma unroll
  for (int r = 0; r < 8; ++r) {
    const size_t mm = (size_t)(m0 + r + hi * 8) * NCOLS;
    ws1[mm + n0 +  0 + lo] = (bf16_t)acc0[r];
    ws1[mm + n0 + 16 + lo] = (bf16_t)acc1[r];
    ws1[mm + n0 + 32 + lo] = (bf16_t)acc2[r];
    ws1[mm + n0 + 48 + lo] = (bf16_t)acc3[r];
  }
}

// ---------------------------------------------------------------------------
// z at the final timestep only: z[b][c] = u[b, L-1] . W[c]  (c < 1536)
// ---------------------------------------------------------------------------
__global__ __launch_bounds__(256) void k_zlast(
    const float* __restrict__ inp, const float* __restrict__ W,
    float* __restrict__ zlast) {
  const int j = blockIdx.x * 256 + threadIdx.x;
  if (j >= NB_ * DINNER) return;
  const int b = j / DINNER, c = j % DINNER;
  const float* __restrict__ urow =
      inp + ((size_t)(b * NT_ + 13) * NTOK + 196) * DMODEL;   // t = L-1
  const float* __restrict__ wrow = W + (size_t)c * DMODEL;
  float acc = 0.f;
#pragma unroll 4
  for (int k = 0; k < DMODEL; ++k) acc = fmaf(urow[k], wrow[k], acc);
  zlast[j] = acc;
}

// ---------------------------------------------------------------------------
// Kernel 2: causal depthwise conv(k=4) + bias + SiLU on xBC columns,
//           and softplus(dt + dt_bias) / dA = exp(-exp(A_log)*dt).
// ---------------------------------------------------------------------------
__global__ __launch_bounds__(256) void k_conv_dt(
    const bf16_t* __restrict__ ws1, const float* __restrict__ conv_w,
    const float* __restrict__ conv_b, const float* __restrict__ dt_bias,
    const float* __restrict__ A_log, float* __restrict__ xconv,
    float* __restrict__ dtdA) {
  const int gid = blockIdx.x * 256 + threadIdx.x;
  const int nconv = MROWS * CONVDIM;
  if (gid < nconv) {
    const int c   = gid % CONVDIM;
    const int row = gid / CONVDIM;              // b*LSEQ + t
    const int b = row / LSEQ, t = row % LSEQ;
    float acc = conv_b[c];
#pragma unroll
    for (int k = 0; k < 4; ++k) {
      const int ts = t - 3 + k;                 // left zero-pad of 3
      if (ts >= 0)
        acc = fmaf(conv_w[c * 4 + k],
                   (float)ws1[(size_t)(b * LSEQ + ts) * NCOLS + c], acc);
    }
    xconv[(size_t)row * CONVDIM + c] = acc / (1.f + __expf(-acc));  // SiLU
  } else {
    const int g2 = gid - nconv;
    if (g2 < MROWS * NHEADS) {
      const int h   = g2 % NHEADS;
      const int row = g2 / NHEADS;
      const float raw =
          (float)ws1[(size_t)row * NCOLS + CONVDIM + h] + dt_bias[h];
      const float dt = raw > 20.f ? raw : log1pf(__expf(raw));      // softplus
      const float dA = __expf(-__expf(A_log[h]) * dt);
      dtdA[((size_t)row * NHEADS + h) * 2 + 0] = dt;
      dtdA[((size_t)row * NHEADS + h) * 2 + 1] = dA;
    }
  }
}

// ---------------------------------------------------------------------------
// Kernel 3: the serial SSM scan. One wave32 per (b, head). State h[8][128]
// lives in 32 VGPRs/lane (n = 4*lane + nb). Only the final y is materialized.
// ---------------------------------------------------------------------------
__global__ __launch_bounds__(128) void k_scan(
    const float* __restrict__ xconv, const float* __restrict__ dtdA,
    const float* __restrict__ Dp, float* __restrict__ yraw) {
  const int lane = threadIdx.x & 31;
  const int wid  = blockIdx.x * 4 + (threadIdx.x >> 5);
  const int b = wid / NHEADS, h = wid % NHEADS;
  const float* __restrict__ xbase = xconv + (size_t)b * LSEQ * CONVDIM;
  const float* __restrict__ dbase = dtdA + (size_t)b * LSEQ * NHEADS * 2;

  float hst[8][4] = {};
  float4 xlo = {}, xhi = {};
#pragma unroll 1
  for (int t = 0; t < LSEQ; ++t) {
    const float* __restrict__ rowp = xbase + (size_t)t * CONVDIM;
    const float4 Bv = *(const float4*)(rowp + DINNER + 4 * lane);   // per-lane
    xlo = *(const float4*)(rowp + h * 8);                           // broadcast
    xhi = *(const float4*)(rowp + h * 8 + 4);
    const float2 dd = *(const float2*)(dbase + ((size_t)t * NHEADS + h) * 2);
    const float dt = dd.x, dA = dd.y;
    if (t + 1 < LSEQ) {                                   // global_prefetch_b8
      __builtin_prefetch(rowp + CONVDIM + DINNER + 4 * lane, 0, 0);
      __builtin_prefetch(rowp + CONVDIM + h * 8, 0, 0);
    }
    const float Bc[4] = {Bv.x, Bv.y, Bv.z, Bv.w};
    const float s[8] = {dt * xlo.x, dt * xlo.y, dt * xlo.z, dt * xlo.w,
                        dt * xhi.x, dt * xhi.y, dt * xhi.z, dt * xhi.w};
#pragma unroll
    for (int p = 0; p < 8; ++p)
#pragma unroll
      for (int nb = 0; nb < 4; ++nb)
        hst[p][nb] = fmaf(hst[p][nb], dA, s[p] * Bc[nb]);
  }
  // y[p] = sum_n h[p][n]*C[n] (+ D*x) at the final step only.
  const float* __restrict__ lastrow = xbase + (size_t)(LSEQ - 1) * CONVDIM;
  const float4 Cv = *(const float4*)(lastrow + DINNER + DSTATE + 4 * lane);
  const float Cc[4] = {Cv.x, Cv.y, Cv.z, Cv.w};
  const float xl[8] = {xlo.x, xlo.y, xlo.z, xlo.w, xhi.x, xhi.y, xhi.z, xhi.w};
  const float Dh = Dp[h];
#pragma unroll
  for (int p = 0; p < 8; ++p) {
    float y = hst[p][0] * Cc[0] + hst[p][1] * Cc[1] +
              hst[p][2] * Cc[2] + hst[p][3] * Cc[3];
#pragma unroll
    for (int off = 16; off >= 1; off >>= 1) y += __shfl_xor(y, off, 32);
    if (lane == 0)
      yraw[(size_t)b * DINNER + h * 8 + p] = y + Dh * xl[p];
  }
}

// ---------------------------------------------------------------------------
// Kernel 4: gate with silu(z), RMSNorm, emit bf16; pad M to 16 rows of zeros.
// ---------------------------------------------------------------------------
__global__ __launch_bounds__(256) void k_norm(
    const float* __restrict__ yraw, const float* __restrict__ zlast,
    const float* __restrict__ norm_w, bf16_t* __restrict__ ynorm) {
  const int b = blockIdx.x, tid = threadIdx.x;
  __shared__ float red[256];
  if (b >= NB_) {                                 // zero-pad rows 4..15
    for (int c = tid; c < DINNER; c += 256)
      ynorm[(size_t)b * DINNER + c] = (bf16_t)0.f;
    return;
  }
  float ss = 0.f;
  for (int c = tid; c < DINNER; c += 256) {
    const float z  = zlast[b * DINNER + c];
    const float yg = yraw[b * DINNER + c] * (z / (1.f + __expf(-z)));
    ss += yg * yg;
  }
  red[tid] = ss;
  __syncthreads();
  for (int st = 128; st > 0; st >>= 1) {
    if (tid < st) red[tid] += red[tid + st];
    __syncthreads();
  }
  const float inv = rsqrtf(red[0] / (float)DINNER + 1e-5f);
  for (int c = tid; c < DINNER; c += 256) {
    const float z  = zlast[b * DINNER + c];
    const float yg = yraw[b * DINNER + c] * (z / (1.f + __expf(-z)));
    ynorm[(size_t)b * DINNER + c] = (bf16_t)(yg * inv * norm_w[c]);
  }
}

// ---------------------------------------------------------------------------
// Kernel 5: classifier GEMM [16 x 1536] @ cls_w^T, add bias, broadcast over T.
// One wave -> one 16x16 tile, 48 k-steps of WMMA bf16.
// ---------------------------------------------------------------------------
__global__ __launch_bounds__(128) void k_cls(
    const bf16_t* __restrict__ ynorm, const float* __restrict__ clsw,
    const float* __restrict__ clsb, float* __restrict__ out) {
  const int lane = threadIdx.x & 31;
  const int wid  = blockIdx.x * 4 + (threadIdx.x >> 5);
  if (wid >= NCLS_TILES) return;
  const int n0 = wid * 16;
  const int lo = lane & 15, hi = lane >> 4;

  const bf16_t* __restrict__ arow = ynorm + (size_t)lo * DINNER;
  const int r = n0 + lo;
  const int rc = r < NUMCLASS ? r : NUMCLASS - 1;           // clamp OOB rows
  const float* __restrict__ brow = clsw + (size_t)rc * DINNER;

  v8f acc = {};
#pragma unroll 4
  for (int ks = 0; ks < DINNER / 32; ++ks) {
    const int kb = ks * 32;
    const v8bf a0 = *(const v8bf*)(arow + kb + hi * 8);
    const v8bf a1 = *(const v8bf*)(arow + kb + 16 + hi * 8);
    v16bf a;
#pragma unroll
    for (int j = 0; j < 8; ++j) { a[j] = a0[j]; a[j + 8] = a1[j]; }
    const v16bf bb = frag_from_f32(brow, kb, hi);
    acc = __builtin_amdgcn_wmma_f32_16x16x32_bf16(false, a, false, bb,
                                                  (short)0, acc, false, false);
  }
  const int col = n0 + lo;
  if (col < NUMCLASS) {
    const float bias = clsb[col];
#pragma unroll
    for (int rr = 0; rr < 8; ++rr) {
      const int mm = rr + hi * 8;
      if (mm < NB_) {
        const float v = acc[rr] + bias;
        for (int t = 0; t < NT_; ++t)
          out[(size_t)(mm * NT_ + t) * NUMCLASS + col] = v;
      }
    }
  }
}

// ---------------------------------------------------------------------------
extern "C" void kernel_launch(void* const* d_in, const int* in_sizes, int n_in,
                              void* d_out, int out_size, void* d_ws,
                              size_t ws_size, hipStream_t stream) {
  (void)in_sizes; (void)n_in; (void)out_size; (void)ws_size;
  const float* inp   = (const float*)d_in[0];
  const float* inw   = (const float*)d_in[1];
  const float* convw = (const float*)d_in[2];
  const float* convb = (const float*)d_in[3];
  const float* dtb   = (const float*)d_in[4];
  const float* alog  = (const float*)d_in[5];
  const float* Dp    = (const float*)d_in[6];
  const float* nw    = (const float*)d_in[7];
  // d_in[8] = out_proj_w is dead code for this output.
  const float* clsw  = (const float*)d_in[9];
  const float* clsb  = (const float*)d_in[10];
  float* out = (float*)d_out;

  char* ws = (char*)d_ws;
  size_t off = 0;
  auto take = [&](size_t bytes) {
    off = (off + 255) & ~(size_t)255;
    void* p = ws + off;
    off += bytes;
    return p;
  };
  bf16_t* ws1   = (bf16_t*)take((size_t)MROWS * NCOLS * sizeof(bf16_t)); // 43.6 MB
  float*  xconv = (float*) take((size_t)MROWS * CONVDIM * sizeof(float)); // 78.7 MB
  float*  dtdA  = (float*) take((size_t)MROWS * NHEADS * 2 * sizeof(float)); // 16.9 MB
  float*  yraw  = (float*) take((size_t)NB_ * DINNER * sizeof(float));
  float*  zlast = (float*) take((size_t)NB_ * DINNER * sizeof(float));
  bf16_t* ynorm = (bf16_t*)take((size_t)16 * DINNER * sizeof(bf16_t));

  // 1) in_proj GEMM (WMMA): 686 * 31 strips, 4 waves / block.
  const int strips = (MROWS / 16) * (NCOLS / 64);
  k_inproj_gemm<<<(strips + 3) / 4, 128, 0, stream>>>(inp, inw, ws1);
  // 1b) z at final timestep.
  k_zlast<<<(NB_ * DINNER + 255) / 256, 256, 0, stream>>>(inp, inw, zlast);
  // 2) conv + SiLU + dt/dA.
  const int tot2 = MROWS * (CONVDIM + NHEADS);
  k_conv_dt<<<(tot2 + 255) / 256, 256, 0, stream>>>(ws1, convw, convb, dtb,
                                                    alog, xconv, dtdA);
  // 3) serial scan: 768 waves, 4 per block.
  k_scan<<<(NB_ * NHEADS) / 4, 128, 0, stream>>>(xconv, dtdA, Dp, yraw);
  // 4) gate + RMSNorm (rows 4..15 zero-padded).
  k_norm<<<16, 256, 0, stream>>>(yraw, zlast, nw, ynorm);
  // 5) classifier GEMM (WMMA) + bias + broadcast over T.
  k_cls<<<(NCLS_TILES + 3) / 4, 128, 0, stream>>>(ynorm, clsw, clsb, out);
}
